// Self_postrans_8684423872636
// MI455X (gfx1250) — matrix operand order repaired and checked by reading, not compiled, not run on hardware
//
#include <hip/hip_runtime.h>
#include <hip/hip_bf16.h>

// ---------------------------------------------------------------------------
// Problem constants (from reference): B=8, N=2048, C=60, DIM=384
// ---------------------------------------------------------------------------
#define BATCH 8
#define NTOK  2048
#define DIMQ  384
#define NCH   60

typedef __attribute__((ext_vector_type(16))) __bf16 v16bf;
typedef __attribute__((ext_vector_type(8)))  __bf16 v8bf;
typedef __attribute__((ext_vector_type(8)))  float  v8f;
typedef __attribute__((ext_vector_type(4)))  int    i32x4;

// Address-space-qualified pointee types for the async-LDS builtin
// (clang prints AS1 as "__device__", AS3 as "__shared__")
typedef __attribute__((address_space(1))) i32x4 gbl_i32x4;
typedef __attribute__((address_space(3))) i32x4 lds_i32x4;

// float -> bf16 (round-to-nearest-even), as raw u16
__device__ __forceinline__ unsigned short f2bfu(float f) {
    unsigned u = __builtin_bit_cast(unsigned, f);
    u += 0x7fffu + ((u >> 16) & 1u);
    return (unsigned short)(u >> 16);
}
__device__ __forceinline__ __bf16 f2bf(float f) {
    unsigned short h = f2bfu(f);
    return __builtin_bit_cast(__bf16, h);
}
// Concatenate two 8x bf16 chunks into one 16x bf16 WMMA operand
__device__ __forceinline__ v16bf join8(const unsigned short* pa,
                                       const unsigned short* pb) {
    v8bf a = *(const v8bf*)pa;
    v8bf b = *(const v8bf*)pb;
    v16bf r;
#pragma unroll
    for (int i = 0; i < 8; ++i) { r[i] = a[i]; r[i + 8] = b[i]; }
    return r;
}

// ---------------------------------------------------------------------------
// Async memory->LDS copy (CDNA5 GLOBAL_LOAD_ASYNC_TO_LDS_B128, ASYNCcnt),
// with a synchronous fallback if the builtin is unavailable.
// ---------------------------------------------------------------------------
#if __has_builtin(__builtin_amdgcn_global_load_async_to_lds_b128)
#define HAVE_ASYNC_LDS 1
#else
#define HAVE_ASYNC_LDS 0
#endif

__device__ __forceinline__ void async_cp16(unsigned short* l,
                                           const unsigned short* g) {
#if HAVE_ASYNC_LDS
    __builtin_amdgcn_global_load_async_to_lds_b128(
        (gbl_i32x4*)g, (lds_i32x4*)l, 0, 0);
#else
    *(uint4*)l = *(const uint4*)g;
#endif
}

__device__ __forceinline__ void wait_async0() {
#if HAVE_ASYNC_LDS
#if __has_builtin(__builtin_amdgcn_s_wait_asynccnt)
    __builtin_amdgcn_s_wait_asynccnt(0);
#else
    asm volatile("s_wait_asynccnt 0x0" ::: "memory");
#endif
#endif
}

// ---------------------------------------------------------------------------
// Stage 1: positional encoding + QKV projection (C=60 -> DIM=384), emit bf16.
//   qT, kT : [B][N][DIM]  (token-major, dim contiguous)  -> WMMA B / A operands
//   vDT    : [B][DIM][N]  (dim-major, token contiguous)  -> WMMA A operand (V^T)
// ---------------------------------------------------------------------------
__global__ __launch_bounds__(384)
void qkv_pe_kernel(const float* __restrict__ pos,
                   const float* __restrict__ Wq, const float* __restrict__ bq,
                   const float* __restrict__ Wk, const float* __restrict__ bk,
                   const float* __restrict__ Wv, const float* __restrict__ bv,
                   unsigned short* __restrict__ qT,
                   unsigned short* __restrict__ kT,
                   unsigned short* __restrict__ vDT)
{
    __shared__ float sPe[8 * NCH];          // 8 tokens x 60 channels
    const int tid  = threadIdx.x;
    const int tok0 = blockIdx.x * 8;

    // phase A: positional encoding for 8 tokens
    for (int idx = tid; idx < 8 * NCH; idx += 384) {
        int tok = idx / NCH, c = idx % NCH;
        int ng  = tok0 + tok;
        int bb  = ng >> 11;                 // / 2048
        int n   = ng & 2047;
        int coord = c / 20, r = c % 20;
        float val = pos[((size_t)(bb * 3 + coord)) * NTOK + n];
        float x   = val * (float)(1 << (r % 10));
        sPe[idx]  = (r < 10) ? __sinf(x) : __cosf(x);
    }
    __syncthreads();

    // phase B: thread d computes q/k/v[d] for the 8 tokens
    const int d = tid;                      // 0..383
    for (int tok = 0; tok < 8; ++tok) {
        int ng = tok0 + tok;
        int bb = ng >> 11, n = ng & 2047;
        float aq = bq[d], ak = bk[d], av = bv[d];
        const float* pe = &sPe[tok * NCH];
#pragma unroll
        for (int c = 0; c < NCH; ++c) {
            float p = pe[c];
            aq = fmaf(Wq[d * NCH + c], p, aq);
            ak = fmaf(Wk[d * NCH + c], p, ak);
            av = fmaf(Wv[d * NCH + c], p, av);
        }
        size_t o = ((size_t)(bb * NTOK + n)) * DIMQ + d;
        qT[o] = f2bfu(aq);
        kT[o] = f2bfu(ak);
        vDT[((size_t)(bb * DIMQ + d)) * NTOK + n] = f2bfu(av);
    }
}

// ---------------------------------------------------------------------------
// Stage 2: flash attention, transposed-score formulation.
//   T = S^T tile:  A = K(16 keys x 32 dims), B = Q^T(32 dims x 16 queries)
//   out^T tile  :  A = V^T(16 dims x 32 keys), B = P^T(32 keys x 16 queries)
// In both C-layouts the lane's column (lane&15) is the QUERY, so online
// softmax stats and accumulator rescale are per-lane + one shfl_xor(16).
// K/V chunks stream through double-buffered LDS via async LDS loads.
// ---------------------------------------------------------------------------
#define KROW 392                            // padded K row stride (halves)
#define VROW 40                             // padded V row stride (halves)

__global__ __launch_bounds__(128)
void attn_kernel(const unsigned short* __restrict__ qT,
                 const unsigned short* __restrict__ kT,
                 const unsigned short* __restrict__ vDT,
                 float* __restrict__ out)
{
    // padded rows -> 16 operand lanes hit 16 distinct banks; double buffered
    __shared__ __align__(16) unsigned short ldsK[2][32 * KROW];
    __shared__ __align__(16) unsigned short ldsV[2][384 * VROW];

    const int tid  = threadIdx.x;
    const int lane = tid & 31;
    const int wid  = tid >> 5;
    const int b    = blockIdx.x >> 5;       // 32 query tiles per batch
    const int qt   = blockIdx.x & 31;
    const int q0   = qt * 64 + wid * 16;    // 16 queries per wave
    const int lm   = lane & 15;             // column (query) index
    const int lh   = lane >> 4;             // wave-half selector

    const unsigned short* kbase = kT  + (size_t)b * NTOK * DIMQ;
    const unsigned short* vbase = vDT + (size_t)b * DIMQ * NTOK;

    // issue chunk-0 fill first so it overlaps the Q register load
    {
#pragma unroll
        for (int it = 0; it < 12; ++it) {
            int u = it * 128 + tid;               // 1536 units of 8 halves
            int row = u / 48, c8 = u % 48;
            async_cp16(&ldsK[0][row * KROW + c8 * 8],
                       &kbase[(size_t)row * DIMQ + c8 * 8]);
        }
#pragma unroll
        for (int it = 0; it < 12; ++it) {
            int u = it * 128 + tid;
            int dd = u >> 2, seg = u & 3;
            async_cp16(&ldsV[0][dd * VROW + seg * 8],
                       &vbase[(size_t)dd * NTOK + seg * 8]);
        }
    }

    // Q resident in registers: 12 B-operand tiles (32 dims each)
    v16bf qreg[12];
    {
        const unsigned short* qp =
            qT + ((size_t)(b * NTOK + q0 + lm)) * DIMQ + lh * 16;
#pragma unroll
        for (int dc = 0; dc < 12; ++dc)
            qreg[dc] = *(const v16bf*)(qp + dc * 32);
    }

    v8f zero = {};
    v8f oacc[24];                           // out^T accumulators, 24 x 16 dims
#pragma unroll
    for (int i = 0; i < 24; ++i) oacc[i] = zero;

    float m_run = -3.0e38f;
    float l_run = 0.0f;
    // attn = (q.k) * sqrt(384)  (reference divides by dim**-0.5); fold log2(e)
    const float SC = 19.595917942265423f * 1.4426950408889634f;

    for (int ch = 0; ch < NTOK / 32; ++ch) {
        const int cur = ch & 1;

        wait_async0();                      // this wave's fill of buf[cur] done
        __syncthreads();                    // all waves' fills done

        // prefetch next chunk into the other buffer (async, overlaps compute)
        if (ch + 1 < NTOK / 32) {
            const int nxt = cur ^ 1;
            const int k1  = (ch + 1) * 32;
            const unsigned short* kg = kbase + (size_t)k1 * DIMQ;
            const unsigned short* vg = vbase + k1;
#pragma unroll
            for (int it = 0; it < 12; ++it) {
                int u = it * 128 + tid;
                int row = u / 48, c8 = u % 48;
                async_cp16(&ldsK[nxt][row * KROW + c8 * 8],
                           &kg[(size_t)row * DIMQ + c8 * 8]);
            }
#pragma unroll
            for (int it = 0; it < 12; ++it) {
                int u = it * 128 + tid;
                int dd = u >> 2, seg = u & 3;
                async_cp16(&ldsV[nxt][dd * VROW + seg * 8],
                           &vg[(size_t)dd * NTOK + seg * 8]);
            }
        }

        const unsigned short* cK = ldsK[cur];
        const unsigned short* cV = ldsV[cur];

        // ---- scores: two 16-key S^T tiles, accumulate over 384 dims ----
        v8f s0 = zero, s1 = zero;
#pragma unroll
        for (int dc = 0; dc < 12; ++dc) {
            const int d0 = dc * 32;
            const unsigned short* kb0 = &cK[lm * KROW + d0 + lh * 16];
            const unsigned short* kb1 = &cK[(16 + lm) * KROW + d0 + lh * 16];
            v16bf a0 = join8(kb0, kb0 + 8);
            v16bf a1 = join8(kb1, kb1 + 8);
            s0 = __builtin_amdgcn_wmma_f32_16x16x32_bf16(
                     false, a0, false, qreg[dc], (short)0, s0, false, false);
            s1 = __builtin_amdgcn_wmma_f32_16x16x32_bf16(
                     false, a1, false, qreg[dc], (short)0, s1, false, false);
        }

        // ---- online softmax (per-lane query; one cross-half shuffle) ----
        float t0[8], t1[8];
        float cm = -3.0e38f;
#pragma unroll
        for (int j = 0; j < 8; ++j) {
            t0[j] = s0[j] * SC; t1[j] = s1[j] * SC;
            cm = fmaxf(cm, fmaxf(t0[j], t1[j]));
        }
        cm = fmaxf(cm, __shfl_xor(cm, 16));
        float m_new = fmaxf(m_run, cm);
        float fac   = exp2f(m_run - m_new);
        float p0[8], p1[8], ls = 0.0f;
#pragma unroll
        for (int j = 0; j < 8; ++j) {
            p0[j] = exp2f(t0[j] - m_new);
            p1[j] = exp2f(t1[j] - m_new);
            ls += p0[j] + p1[j];
        }
        ls += __shfl_xor(ls, 16);
        l_run = l_run * fac + ls;
        m_run = m_new;

        // ---- build P^T B-operand (32 keys x 16 queries) via half swaps ----
        v16bf pB;
        {
            const bool low = (lh == 0);
#pragma unroll
            for (int j = 0; j < 8; ++j) {
                float o0 = __shfl_xor(p0[j], 16);
                float o1 = __shfl_xor(p1[j], 16);
                pB[j]     = f2bf(low ? p0[j] : o1);
                pB[j + 8] = f2bf(low ? o0 : p1[j]);
            }
        }

        // ---- PV: out^T += V^T x P^T over 24 dim tiles ----
#pragma unroll
        for (int dt = 0; dt < 24; ++dt) {
            const unsigned short* vb = &cV[(dt * 16 + lm) * VROW + lh * 16];
            v16bf av = join8(vb, vb + 8);
            oacc[dt] = oacc[dt] * fac;
            oacc[dt] = __builtin_amdgcn_wmma_f32_16x16x32_bf16(
                           false, av, false, pB, (short)0, oacc[dt], false, false);
        }
        __syncthreads();                    // compute done before buf reuse
    }

    // ---- epilogue: out[b, dim, n] = acc / l ----
    const float inv = 1.0f / l_run;
    const int   qcol = q0 + lm;
#pragma unroll
    for (int dt = 0; dt < 24; ++dt) {
#pragma unroll
        for (int r = 0; r < 8; ++r) {
            int dim = dt * 16 + r + lh * 8;
            out[((size_t)(b * DIMQ + dim)) * NTOK + qcol] = oacc[dt][r] * inv;
        }
    }
}

// ---------------------------------------------------------------------------
extern "C" void kernel_launch(void* const* d_in, const int* in_sizes, int n_in,
                              void* d_out, int out_size, void* d_ws, size_t ws_size,
                              hipStream_t stream) {
    (void)in_sizes; (void)n_in; (void)out_size; (void)ws_size;
    const float* pos = (const float*)d_in[0];
    const float* Wq  = (const float*)d_in[1];
    const float* bq  = (const float*)d_in[2];
    const float* Wk  = (const float*)d_in[3];
    const float* bk  = (const float*)d_in[4];
    const float* Wv  = (const float*)d_in[5];
    const float* bv  = (const float*)d_in[6];
    float* out = (float*)d_out;

    const size_t QKV = (size_t)BATCH * NTOK * DIMQ;   // elems per matrix
    unsigned short* ws  = (unsigned short*)d_ws;
    unsigned short* qT  = ws;
    unsigned short* kT  = ws + QKV;
    unsigned short* vDT = ws + 2 * QKV;

    // Stage 1: 2048 blocks x 384 threads, 8 tokens/block
    qkv_pe_kernel<<<BATCH * NTOK / 8, 384, 0, stream>>>(
        pos, Wq, bq, Wk, bk, Wv, bv, qT, kT, vDT);

    // Stage 2: flash attention; 8 batches x 32 query tiles, 4 waves/WG
    attn_kernel<<<BATCH * (NTOK / 64), 128, 0, stream>>>(qT, kT, vDT, out);
}
